// PAConv_42099269435606
// MI455X (gfx1250) — matrix operand order repaired and checked by reading, not compiled, not run on hardware
//
#include <hip/hip_runtime.h>
#include <hip/hip_bf16.h>
#include <math.h>

// ---------------------------------------------------------------------------
// MI455X (gfx1250) fused point-transformer layer.
//   - KNN (brute force, top-32 per query)
//   - kh = [G | relu(rel@Wp1+bp1)] @ [Wk;Wp2] + (bk+bp2)   (single K=192 WMMA GEMM)
//   - v  =  G @ Wv + bv                                    (K=64 WMMA GEMM)
//   - 4-head attention over 32 neighbors, softmax in fp32
//   - training-mode BatchNorm, deterministic 2-stage reduction
// WMMA: v_wmma_f32_16x16x32_f16 (f16 operands, f32 accum).
// ---------------------------------------------------------------------------

#define B_SZ   8
#define N_PTS  4096
#define C_IN   64
#define C_OUT  128
#define KNN_K  32
#define HEADS  4
#define HDIM   32
#define KAUG   192          // 64 (features) + 128 (pos hidden)
#define ASTR   200          // LDS row stride for A tiles (f16), 16B aligned
#define ROWS_TOTAL (B_SZ * N_PTS)   // 32768

typedef _Float16 f16;
typedef __attribute__((ext_vector_type(16))) _Float16 v16h;
typedef __attribute__((ext_vector_type(8)))  _Float16 v8h;
typedef __attribute__((ext_vector_type(8)))  float    v8f;

// ---------------------------------------------------------------------------
// Kernel 1: brute-force KNN.  grid (B, N/256) x 256 threads, 1 query/thread.
// ---------------------------------------------------------------------------
__global__ __launch_bounds__(256) void knn_kernel(const float* __restrict__ xyz,
                                                  int* __restrict__ idx_out) {
    const int b = blockIdx.x;
    const int n = blockIdx.y * 256 + threadIdx.x;
    const float px = xyz[(b * N_PTS + n) * 3 + 0];
    const float py = xyz[(b * N_PTS + n) * 3 + 1];
    const float pz = xyz[(b * N_PTS + n) * 3 + 2];

    float bd[KNN_K];
    int   bi[KNN_K];
#pragma unroll
    for (int k = 0; k < KNN_K; ++k) { bd[k] = 3.0e38f; bi[k] = 0; }

    __shared__ float sx[256], sy[256], sz[256];

    for (int t = 0; t < N_PTS; t += 256) {
        const int c = t + threadIdx.x;
        sx[threadIdx.x] = xyz[(b * N_PTS + c) * 3 + 0];
        sy[threadIdx.x] = xyz[(b * N_PTS + c) * 3 + 1];
        sz[threadIdx.x] = xyz[(b * N_PTS + c) * 3 + 2];
        __syncthreads();
        for (int j = 0; j < 256; ++j) {
            const float dx = px - sx[j];
            const float dy = py - sy[j];
            const float dz = pz - sz[j];
            const float d = dx * dx + dy * dy + dz * dz;
            if (d < bd[KNN_K - 1]) {
                int pos = KNN_K - 1;
                while (pos > 0 && bd[pos - 1] > d) {
                    bd[pos] = bd[pos - 1];
                    bi[pos] = bi[pos - 1];
                    --pos;
                }
                bd[pos] = d;
                bi[pos] = t + j;
            }
        }
        __syncthreads();
    }
#pragma unroll
    for (int k = 0; k < KNN_K; ++k)
        idx_out[(b * N_PTS + n) * KNN_K + k] = bi[k];
}

// ---------------------------------------------------------------------------
// Kernel 2: weight prep.  Build f16 column-major stacked weights:
//   Wkh_cm[c][0:192] = { Wk[:,c] ; Wp2[:,c] }   (per column, K contiguous)
//   Wv_cm [c][0:64]  =  Wv[:,c]
//   bkh[c] = bk[c] + bp2[c]
// Column-major K-contiguous layout matches the contiguous WMMA B k order used here
// (lane holds 16 contiguous K values of one column) -> two b128 loads/lane.
// ---------------------------------------------------------------------------
__global__ __launch_bounds__(256) void prep_kernel(const float* __restrict__ Wk,
                                                   const float* __restrict__ Wv,
                                                   const float* __restrict__ Wp2,
                                                   const float* __restrict__ bk,
                                                   const float* __restrict__ bp2,
                                                   f16* __restrict__ wkh_cm,
                                                   f16* __restrict__ wv_cm,
                                                   float* __restrict__ bkh) {
    const int id = blockIdx.x * 256 + threadIdx.x;
    if (id < KAUG * C_OUT) {
        const int kk = id / C_OUT, c = id % C_OUT;
        const float w = (kk < C_IN) ? Wk[kk * C_OUT + c] : Wp2[(kk - C_IN) * C_OUT + c];
        wkh_cm[c * KAUG + kk] = (f16)w;
    } else if (id < KAUG * C_OUT + C_IN * C_OUT) {
        const int i = id - KAUG * C_OUT;
        const int k = i / C_OUT, c = i % C_OUT;
        wv_cm[c * C_IN + k] = (f16)Wv[k * C_OUT + c];
    } else if (id < KAUG * C_OUT + C_IN * C_OUT + C_OUT) {
        const int c = id - (KAUG * C_OUT + C_IN * C_OUT);
        bkh[c] = bk[c] + bp2[c];
    }
}

// ---------------------------------------------------------------------------
// Kernel 3: fused gather + pos MLP + WMMA GEMMs + attention.
// One block = 16 queries (same batch).  8 waves; wave w owns output columns
// [16w, 16w+16) and keeps its B-slices resident in 8 x v16h VGPRs.
// Chunk = 1 query x 32 neighbor rows = 2 WMMA M-tiles.
// ---------------------------------------------------------------------------
__global__ __launch_bounds__(256, 1) void fused_attn_kernel(
    const float* __restrict__ xyz, const float* __restrict__ points,
    const float* __restrict__ Wq, const float* __restrict__ bq,
    const float* __restrict__ Wp1, const float* __restrict__ bp1,
    const float* __restrict__ bv,
    const f16* __restrict__ wkh_cm, const f16* __restrict__ wv_cm,
    const float* __restrict__ bkh,
    const int* __restrict__ idx, float* __restrict__ out) {
    __shared__ f16   sA[32][ASTR];        // augmented A chunk [G | pos-hidden]
    __shared__ float sK[32][C_OUT];       // kh for current chunk (fp32)
    __shared__ float sV[32][C_OUT];       // v  for current chunk (fp32)
    __shared__ float sQ[16][C_OUT];       // q for the block's 16 queries
    __shared__ float sPts[16][C_IN];      // query feature rows
    __shared__ float sQX[16][3];          // query coords
    __shared__ int   sIdx[KNN_K];
    __shared__ float sRel[KNN_K][3];
    __shared__ float sS[KNN_K][HEADS];    // attention scores / weights

    const int t    = threadIdx.x;
    const int wave = t >> 5;
    const int lane = t & 31;
    const int blk  = blockIdx.x;
    const int b    = (blk * 16) / N_PTS;
    const int n0   = (blk * 16) % N_PTS;

    // ---- preamble: stage query rows, compute q = pts @ Wq + bq (VALU, ~1.5% of FLOPs)
    for (int i = t; i < 16 * C_IN; i += 256)
        sPts[i >> 6][i & 63] = points[(b * N_PTS + n0 + (i >> 6)) * C_IN + (i & 63)];
    if (t < 16) {
#pragma unroll
        for (int d = 0; d < 3; ++d)
            sQX[t][d] = xyz[(b * N_PTS + n0 + t) * 3 + d];
    }
    __syncthreads();
    for (int o = t; o < 16 * C_OUT; o += 256) {
        const int qi = o >> 7, c = o & 127;
        float acc = bq[c];
#pragma unroll 8
        for (int k = 0; k < C_IN; ++k) acc += sPts[qi][k] * Wq[k * C_OUT + c];
        sQ[qi][c] = acc;
    }

    // ---- per-wave resident B operands (WMMA 16-bit B lane layout)
    const int col = wave * 16 + (lane & 15);   // output column owned by this lane
    const int kb  = (lane >> 4) * 16;          // K sub-range within 32-K step
    v16h Bk[6], Bv2[2];
#pragma unroll
    for (int s = 0; s < 6; ++s)
        Bk[s] = *(const v16h*)(wkh_cm + col * KAUG + s * 32 + kb);
#pragma unroll
    for (int s = 0; s < 2; ++s)
        Bv2[s] = *(const v16h*)(wv_cm + col * C_IN + s * 32 + kb);
    const float biask = bkh[col];
    const float biasv = bv[col];
    __syncthreads();

    // ---- chunk loop: one query per chunk
    for (int qi = 0; qi < 16; ++qi) {
        const int gq = n0 + qi;
        if (t < KNN_K) sIdx[t] = idx[(b * N_PTS + gq) * KNN_K + t];
        if (qi + 1 < 16)
            __builtin_prefetch(&idx[(b * N_PTS + gq + 1) * KNN_K], 0, 0);
        __syncthreads();

        // gather neighbor features -> sA[:,0:64] (f16)
        for (int e = t; e < KNN_K * C_IN; e += 256) {
            const int r = e >> 6, c = e & 63;
            sA[r][c] = (f16)points[(b * N_PTS + sIdx[r]) * C_IN + c];
        }
        if (t < KNN_K) {
            const int j = sIdx[t];
#pragma unroll
            for (int d = 0; d < 3; ++d)
                sRel[t][d] = xyz[(b * N_PTS + j) * 3 + d] - sQX[qi][d];
        }
        __syncthreads();

        // pos-MLP layer 1 (K=3, VALU) -> sA[:,64:192] (f16)
        for (int o = t; o < KNN_K * C_OUT; o += 256) {
            const int r = o >> 7, c = o & 127;
            float h = sRel[r][0] * Wp1[c] + sRel[r][1] * Wp1[C_OUT + c] +
                      sRel[r][2] * Wp1[2 * C_OUT + c] + bp1[c];
            sA[r][C_IN + c] = (f16)(h > 0.0f ? h : 0.0f);
        }
        __syncthreads();

        // ---- WMMA GEMMs: kh (K=192, 6 steps) and v (K=64, 2 steps), 2 M-tiles
        {
            const int arow = lane & 15;
            const int ao   = (lane >> 4) * 16;
            v8f accK0 = {}, accK1 = {}, accV0 = {}, accV1 = {};
#pragma unroll
            for (int s = 0; s < 6; ++s) {
                const f16* p0 = &sA[arow][s * 32 + ao];
                const f16* p1 = &sA[16 + arow][s * 32 + ao];
                const v8h lo0 = *(const v8h*)p0;
                const v8h hi0 = *(const v8h*)(p0 + 8);
                const v8h lo1 = *(const v8h*)p1;
                const v8h hi1 = *(const v8h*)(p1 + 8);
                v16h a0, a1;
#pragma unroll
                for (int i = 0; i < 8; ++i) {
                    a0[i] = lo0[i]; a0[8 + i] = hi0[i];
                    a1[i] = lo1[i]; a1[8 + i] = hi1[i];
                }
                accK0 = __builtin_amdgcn_wmma_f32_16x16x32_f16(
                    false, a0, false, Bk[s], (short)0, accK0, false, false);
                accK1 = __builtin_amdgcn_wmma_f32_16x16x32_f16(
                    false, a1, false, Bk[s], (short)0, accK1, false, false);
                if (s < 2) {
                    accV0 = __builtin_amdgcn_wmma_f32_16x16x32_f16(
                        false, a0, false, Bv2[s], (short)0, accV0, false, false);
                    accV1 = __builtin_amdgcn_wmma_f32_16x16x32_f16(
                        false, a1, false, Bv2[s], (short)0, accV1, false, false);
                }
            }
            // D layout: VGPR r, row = (lane>>4)*8 + r, col = 16w + (lane&15)
            const int crow = (lane >> 4) * 8;
#pragma unroll
            for (int r = 0; r < 8; ++r) {
                sK[crow + r][col]      = accK0[r] + biask;
                sK[16 + crow + r][col] = accK1[r] + biask;
                sV[crow + r][col]      = accV0[r] + biasv;
                sV[16 + crow + r][col] = accV1[r] + biasv;
            }
        }
        __syncthreads();

        // ---- attention scores (fp32): attn[n,h] = q_h . kh_h / sqrt(32)
        if (t < KNN_K * HEADS) {
            const int nn = t >> 2, h = t & 3;
            float a = 0.0f;
#pragma unroll 8
            for (int d = 0; d < HDIM; ++d)
                a += sQ[qi][h * HDIM + d] * sK[nn][h * HDIM + d];
            sS[nn][h] = a * 0.17677669529663687f;   // 1/sqrt(32)
        }
        __syncthreads();
        if (t < HEADS) {
            float m = -3.0e38f;
            for (int n = 0; n < KNN_K; ++n) m = fmaxf(m, sS[n][t]);
            float sum = 0.0f;
            for (int n = 0; n < KNN_K; ++n) {
                const float e = __expf(sS[n][t] - m);
                sS[n][t] = e;
                sum += e;
            }
            const float inv = 1.0f / sum;
            for (int n = 0; n < KNN_K; ++n) sS[n][t] *= inv;
        }
        __syncthreads();
        // ---- weighted sum over neighbors -> pre-BN output
        if (t < C_OUT) {
            const int c = t, h = c >> 5;
            float o = 0.0f;
#pragma unroll 8
            for (int n = 0; n < KNN_K; ++n) o += sS[n][h] * sV[n][c];
            out[(b * N_PTS + gq) * C_OUT + c] = o;
        }
        __syncthreads();   // protect sA/sK/sV before next chunk
    }
}

// ---------------------------------------------------------------------------
// Kernels 4-6: deterministic BatchNorm (2-stage reduction, no float atomics).
// ---------------------------------------------------------------------------
__global__ __launch_bounds__(128) void bn_stats1_kernel(const float* __restrict__ out,
                                                        float* __restrict__ psum,
                                                        float* __restrict__ psum2) {
    const int blk = blockIdx.x;        // 256 blocks x 128 rows
    const int c   = threadIdx.x;       // channel
    const int r0  = blk * 128;
    float s = 0.0f, s2 = 0.0f;
    for (int r = 0; r < 128; ++r) {
        const float x = out[(r0 + r) * C_OUT + c];
        s += x;
        s2 += x * x;
    }
    psum[blk * C_OUT + c]  = s;
    psum2[blk * C_OUT + c] = s2;
}

__global__ __launch_bounds__(128) void bn_stats2_kernel(const float* __restrict__ psum,
                                                        const float* __restrict__ psum2,
                                                        const float* __restrict__ gamma,
                                                        const float* __restrict__ beta,
                                                        float* __restrict__ ss) {
    const int c = threadIdx.x;
    float s = 0.0f, s2 = 0.0f;
    for (int b = 0; b < 256; ++b) {
        s += psum[b * C_OUT + c];
        s2 += psum2[b * C_OUT + c];
    }
    const float inv  = 1.0f / (float)ROWS_TOTAL;
    const float mean = s * inv;
    const float var  = s2 * inv - mean * mean;    // biased variance (jnp.var)
    const float sc   = gamma[c] * rsqrtf(var + 1e-5f);
    ss[c]          = sc;
    ss[C_OUT + c]  = beta[c] - mean * sc;
}

__global__ __launch_bounds__(256) void bn_apply_kernel(float* __restrict__ out,
                                                       const float* __restrict__ ss) {
    const int i = blockIdx.x * 256 + threadIdx.x;
    const int c = i & (C_OUT - 1);
    out[i] = out[i] * ss[c] + ss[C_OUT + c];
}

// ---------------------------------------------------------------------------
// Launcher
// ---------------------------------------------------------------------------
extern "C" void kernel_launch(void* const* d_in, const int* in_sizes, int n_in,
                              void* d_out, int out_size, void* d_ws, size_t ws_size,
                              hipStream_t stream) {
    const float* xyz    = (const float*)d_in[0];
    const float* points = (const float*)d_in[1];
    const float* Wq     = (const float*)d_in[2];
    const float* bq     = (const float*)d_in[3];
    const float* Wk     = (const float*)d_in[4];
    const float* bk     = (const float*)d_in[5];
    const float* Wv     = (const float*)d_in[6];
    const float* bv     = (const float*)d_in[7];
    const float* Wp1    = (const float*)d_in[8];
    const float* bp1    = (const float*)d_in[9];
    const float* Wp2    = (const float*)d_in[10];
    const float* bp2    = (const float*)d_in[11];
    const float* gamma  = (const float*)d_in[12];
    const float* beta   = (const float*)d_in[13];
    float* out = (float*)d_out;

    // workspace layout (all offsets 32B-aligned)
    char* w = (char*)d_ws;
    int*   ws_idx = (int*)w;                                   // 4,194,304 B
    f16*   ws_wkh = (f16*)(w + 4194304);                       //    49,152 B
    f16*   ws_wv  = (f16*)(w + 4194304 + 49152);               //    16,384 B
    float* ws_bkh = (float*)(w + 4194304 + 65536);             //       512 B
    float* ws_ps  = (float*)(w + 4194304 + 66048);             //   131,072 B
    float* ws_ps2 = (float*)(w + 4194304 + 66048 + 131072);    //   131,072 B
    float* ws_ss  = (float*)(w + 4194304 + 66048 + 262144);    //     1,024 B

    knn_kernel<<<dim3(B_SZ, N_PTS / 256), 256, 0, stream>>>(xyz, ws_idx);

    const int prep_elems = KAUG * C_OUT + C_IN * C_OUT + C_OUT;
    prep_kernel<<<(prep_elems + 255) / 256, 256, 0, stream>>>(
        Wk, Wv, Wp2, bk, bp2, ws_wkh, ws_wv, ws_bkh);

    fused_attn_kernel<<<ROWS_TOTAL / 16, 256, 0, stream>>>(
        xyz, points, Wq, bq, Wp1, bp1, bv, ws_wkh, ws_wv, ws_bkh, ws_idx, out);

    bn_stats1_kernel<<<256, 128, 0, stream>>>(out, ws_ps, ws_ps2);
    bn_stats2_kernel<<<1, 128, 0, stream>>>(ws_ps, ws_ps2, gamma, beta, ws_ss);
    bn_apply_kernel<<<(ROWS_TOTAL * C_OUT) / 256, 256, 0, stream>>>(out, ws_ss);
}
